// PerceiverAttention_78829829750858
// MI455X (gfx1250) — compile-verified, matrix operand order checked by reading, not run on hardware
//
#include <hip/hip_runtime.h>

// ---------------------------------------------------------------------------
// Types (clang ext_vectors; HIP class-vectors can't live in unions)
// ---------------------------------------------------------------------------
typedef __attribute__((ext_vector_type(16))) __bf16       v16bf;
typedef __attribute__((ext_vector_type(8)))  float        v8f;
typedef __attribute__((ext_vector_type(4)))  unsigned int u32x4;
typedef __attribute__((ext_vector_type(2)))  unsigned int u32x2;

union FragBf { v16bf v; u32x4 q[2]; };

// ---------------------------------------------------------------------------
// Problem constants
// ---------------------------------------------------------------------------
#define BATCH   8
#define MSEQ    4096
#define NLAT    64
#define DIM     1024
#define HEADS   16
#define DHEAD   64
#define SKV     4160                 // MSEQ + NLAT
#define MKV     (BATCH * SKV)        // 33280 rows in the big GEMM
#define NSPLIT  5                    // s-dim split for attention (5 * 832)
#define SCHUNK  832                  // 13 * 64
#define SOFT_SCALE (1.0f / 64.0f)    // (d^-0.5)^2

// ---------------------------------------------------------------------------
// WMMA + async helpers
// ---------------------------------------------------------------------------
__device__ __forceinline__ v8f wmma_bf16(v16bf a, v16bf b, v8f c) {
  return __builtin_amdgcn_wmma_f32_16x16x32_bf16(false, a, false, b,
                                                 (short)0, c, false, false);
}

__device__ __forceinline__ v8f zero8() {
  v8f z;
#pragma unroll
  for (int i = 0; i < 8; ++i) z[i] = 0.0f;
  return z;
}

// Async copy of 16 bytes global -> LDS (per-lane), tracked by ASYNCcnt.
// VDST = VGPR holding the LDS byte address, then 64-bit global address.
__device__ __forceinline__ void async_copy_b128(unsigned lds_off,
                                                const __bf16* gsrc) {
  asm volatile("global_load_async_to_lds_b128 %0, %1, off"
               :: "v"(lds_off), "v"((unsigned long long)(size_t)gsrc)
               : "memory");
}

__device__ __forceinline__ void wait_async_le4() {
  asm volatile("s_wait_asynccnt 0x4" ::: "memory");
}
__device__ __forceinline__ void wait_async_0() {
  asm volatile("s_wait_asynccnt 0x0" ::: "memory");
}

// Fragment loader for 16-bit A-layout (ISA 7.12.2): lane l<16 holds row
// row0+l, K {k0..k0+7} and {k0+16..k0+23}; lanes 16..31 hold the same rows
// with K {k0+8..k0+15} and {k0+24..k0+31}.  Works for global row-major
// matrices, transposed weights, and LDS tiles (generic pointer -> ds_load).
__device__ __forceinline__ v16bf load_frag(const __bf16* base, int ld,
                                           int row0, int k0) {
  const int l  = threadIdx.x & 31;
  const int r  = row0 + (l & 15);
  const int kk = k0 + ((l & 16) ? 8 : 0);
  const __bf16* p = base + (size_t)r * ld + kk;
  FragBf f;
  f.q[0] = *reinterpret_cast<const u32x4*>(p);
  f.q[1] = *reinterpret_cast<const u32x4*>(p + 16);
  return f.v;
}

// ---------------------------------------------------------------------------
// Kernel 1: fused LayerNorm + f32->bf16 into concat kv buffer [B, SKV, DIM]
// ---------------------------------------------------------------------------
__global__ void __launch_bounds__(256)
ln_to_bf16_kernel(const float* __restrict__ in, const float* __restrict__ g,
                  const float* __restrict__ be, __bf16* __restrict__ out,
                  int rowsPerBatch, int rowOffset) {
  __shared__ float ssum[256];
  __shared__ float ssq[256];
  const int row = blockIdx.x;
  const int t   = threadIdx.x;

  const float4 v = reinterpret_cast<const float4*>(in + (size_t)row * DIM)[t];
  ssum[t] = v.x + v.y + v.z + v.w;
  ssq[t]  = v.x * v.x + v.y * v.y + v.z * v.z + v.w * v.w;
  __syncthreads();
#pragma unroll
  for (int off = 128; off > 0; off >>= 1) {
    if (t < off) { ssum[t] += ssum[t + off]; ssq[t] += ssq[t + off]; }
    __syncthreads();
  }
  const float mu   = ssum[0] * (1.0f / DIM);
  const float var  = ssq[0] * (1.0f / DIM) - mu * mu;
  const float rstd = rsqrtf(var + 1e-5f);

  const int bI = row / rowsPerBatch;
  const int r  = row - bI * rowsPerBatch;
  __bf16* op = out + ((size_t)bI * SKV + rowOffset + r) * DIM + t * 4;

  const float vv[4] = {v.x, v.y, v.z, v.w};
  union { __bf16 h[4]; u32x2 u; } pk;
#pragma unroll
  for (int i = 0; i < 4; ++i) {
    const int c = t * 4 + i;
    pk.h[i] = (__bf16)((vv[i] - mu) * rstd * g[c] + be[c]);
  }
  *reinterpret_cast<u32x2*>(op) = pk.u;
}

// ---------------------------------------------------------------------------
// Kernel 2: weight transpose + convert: W[K,N] f32 -> WT[N,K] bf16
// ---------------------------------------------------------------------------
__global__ void __launch_bounds__(256)
transpose_to_bf16_kernel(const float* __restrict__ in, __bf16* __restrict__ out,
                         int K, int N) {
  __shared__ float tile[32][33];
  const int n0 = blockIdx.x * 32;
  const int k0 = blockIdx.y * 32;
  const int tx = threadIdx.x;   // 0..31
  const int ty = threadIdx.y;   // 0..7
#pragma unroll
  for (int i = 0; i < 32; i += 8)
    tile[ty + i][tx] = in[(size_t)(k0 + ty + i) * N + n0 + tx];
  __syncthreads();
#pragma unroll
  for (int i = 0; i < 32; i += 8)
    out[(size_t)(n0 + ty + i) * K + k0 + tx] = (__bf16)tile[tx][ty + i];
}

// ---------------------------------------------------------------------------
// Kernel 3: WMMA bf16 GEMM with double-buffered async-to-LDS staging.
// C[M,N] = A[M,K] * BT[N,K]^T.  256 threads = 8 waves, block tile 128x128,
// wave tile 32x64, K step 32.  LDS: 2 x (128x32 A + 128x32 B) bf16 = 32 KB.
// MODE 0: bf16 row-major -> C0 (ldc)
// MODE 1: f32  row-major -> C0 (ldc)
// MODE 2: kv epilogue: n<1024 -> K[b,s,1024]; n>=1024 -> Vt[b,1024,SKV]
// ---------------------------------------------------------------------------
template <int MODE>
__global__ void __launch_bounds__(256)
gemm_wmma_kernel(const __bf16* __restrict__ A, const __bf16* __restrict__ BT,
                 int K, int lda, int rpbA, size_t bstrideA,
                 void* __restrict__ C0, void* __restrict__ C1, int ldc) {
  __shared__ __align__(16) __bf16 sA[2][128 * 32];
  __shared__ __align__(16) __bf16 sB[2][128 * 32];

  const int tid  = threadIdx.x;
  const int lane = tid & 31;
  const int wave = tid >> 5;
  const int wm = wave >> 1;                  // 0..3
  const int wn = wave & 1;                   // 0..1
  const int blkRow = blockIdx.x * 128;
  const int blkCol = blockIdx.y * 128;

  // Cooperative stage mapping: 512 x 16B transfers per operand tile;
  // this thread covers rows r0 and r0+64, segment sg (8 bf16 = 16B).
  const int r0 = tid >> 2;                   // 0..63
  const int sg = tid & 3;                    // 0..3
  const unsigned ldsA = (unsigned)(size_t)(&sA[0][0]);
  const unsigned ldsB = (unsigned)(size_t)(&sB[0][0]);
  const unsigned ldOff = (unsigned)((r0 * 32 + sg * 8) * 2);

  auto a_row_off = [&](int r) -> size_t {
    if (bstrideA) {
      const int bb = r / rpbA;
      return (size_t)bb * bstrideA + (size_t)(r - bb * rpbA) * lda;
    }
    return (size_t)r * lda;
  };

  auto stage = [&](int buf, int kk) {
    const unsigned la = ldsA + (unsigned)buf * 128 * 32 * 2 + ldOff;
    const unsigned lb = ldsB + (unsigned)buf * 128 * 32 * 2 + ldOff;
    async_copy_b128(la,                A  + a_row_off(blkRow + r0)      + kk + sg * 8);
    async_copy_b128(la + 64 * 32 * 2,  A  + a_row_off(blkRow + 64 + r0) + kk + sg * 8);
    async_copy_b128(lb,                BT + (size_t)(blkCol + r0) * K      + kk + sg * 8);
    async_copy_b128(lb + 64 * 32 * 2,  BT + (size_t)(blkCol + 64 + r0) * K + kk + sg * 8);
  };

  v8f acc[2][4];
#pragma unroll
  for (int i = 0; i < 2; ++i)
#pragma unroll
    for (int j = 0; j < 4; ++j) acc[i][j] = zero8();

  stage(0, 0);
  int buf = 0;
  for (int kk = 0; kk < K; kk += 32) {
    if (kk + 32 < K) {
      stage(buf ^ 1, kk + 32);
      wait_async_le4();                   // our 4 oldest (= buf) are done
    } else {
      wait_async_0();
    }
    __syncthreads();                      // all waves' buf transfers visible

    const __bf16* pA = &sA[buf][0];
    const __bf16* pB = &sB[buf][0];
    const v16bf a0 = load_frag(pA, 32, wm * 32,      0);
    const v16bf a1 = load_frag(pA, 32, wm * 32 + 16, 0);
#pragma unroll
    for (int j = 0; j < 4; ++j) {
      const v16bf bb = load_frag(pB, 32, wn * 64 + j * 16, 0);
      acc[0][j] = wmma_bf16(a0, bb, acc[0][j]);
      acc[1][j] = wmma_bf16(a1, bb, acc[1][j]);
    }
    __syncthreads();                      // done reading buf before refill
    buf ^= 1;
  }

  // Epilogue.  C layout: VGPR r, lanes 0-15 -> M=r, N=lane; lanes 16-31 ->
  // M=r+8, N=lane-16.
  const int row0 = blkRow + wm * 32;
  const int col0 = blkCol + wn * 64;
#pragma unroll
  for (int i = 0; i < 2; ++i)
#pragma unroll
    for (int j = 0; j < 4; ++j)
#pragma unroll
      for (int r = 0; r < 8; ++r) {
        const int m = row0 + i * 16 + r + ((lane & 16) >> 1);
        const int n = col0 + j * 16 + (lane & 15);
        const float v = acc[i][j][r];
        if (MODE == 0) {
          ((__bf16*)C0)[(size_t)m * ldc + n] = (__bf16)v;
        } else if (MODE == 1) {
          ((float*)C0)[(size_t)m * ldc + n] = v;
        } else {
          const int bb = m / SKV;
          const int s  = m - bb * SKV;
          if (n < DIM) {   // K half: row-major [b, s, DIM]
            ((__bf16*)C0)[((size_t)bb * SKV + s) * DIM + n] = (__bf16)v;
          } else {         // V half: transposed [b, DIM, SKV]
            ((__bf16*)C1)[((size_t)bb * DIM + (n - DIM)) * SKV + s] = (__bf16)v;
          }
        }
      }
}

// ---------------------------------------------------------------------------
// Kernel 4: flash attention partials.  Grid (HEADS, BATCH, NSPLIT); each
// block handles 832 s-positions in chunks of 64.  128 threads = 4 waves,
// one wave per 16 query rows.  Writes unnormalized O + per-row (m, l).
// ---------------------------------------------------------------------------
__global__ void __launch_bounds__(128)
attn_wmma_kernel(const __bf16* __restrict__ Q, const __bf16* __restrict__ Kb,
                 const __bf16* __restrict__ Vt, float* __restrict__ Opart,
                 float* __restrict__ mlbuf) {
  __shared__ __align__(16) __bf16 smem[4 * 16 * 72];  // P tiles, padded
  const int h = blockIdx.x;
  const int b = blockIdx.y;
  const int split = blockIdx.z;
  const int lane = threadIdx.x & 31;
  const int wave = threadIdx.x >> 5;
  const int row0 = wave * 16;
  const int sBeg = split * SCHUNK;

  const __bf16* Qp = Q  + (size_t)b * NLAT * DIM + h * DHEAD;
  const __bf16* Kp = Kb + (size_t)b * SKV * DIM + h * DHEAD;
  const __bf16* Vp = Vt + (size_t)b * DIM * SKV + (size_t)h * DHEAD * SKV;
  __bf16* Pl = smem + wave * 16 * 72;

  const v16bf aq0 = load_frag(Qp, DIM, row0, 0);
  const v16bf aq1 = load_frag(Qp, DIM, row0, 32);

  v8f o[4];
#pragma unroll
  for (int j = 0; j < 4; ++j) o[j] = zero8();
  float mr[8], lr[8];
#pragma unroll
  for (int r = 0; r < 8; ++r) { mr[r] = -1e30f; lr[r] = 0.0f; }

  for (int c = 0; c < SCHUNK / 64; ++c) {
    const int s0 = sBeg + c * 64;
    // ---- S = (Q K^T) * scale, 16x64 f32 per wave ----
    v8f s[4];
#pragma unroll
    for (int j = 0; j < 4; ++j) {
      s[j] = zero8();
      v16bf bk = load_frag(Kp, DIM, s0 + j * 16, 0);
      s[j] = wmma_bf16(aq0, bk, s[j]);
      bk = load_frag(Kp, DIM, s0 + j * 16, 32);
      s[j] = wmma_bf16(aq1, bk, s[j]);
    }
#pragma unroll
    for (int j = 0; j < 4; ++j)
#pragma unroll
      for (int r = 0; r < 8; ++r) s[j][r] *= SOFT_SCALE;

    // ---- online softmax (16-lane row reductions) ----
    float cm[8];
#pragma unroll
    for (int r = 0; r < 8; ++r)
      cm[r] = fmaxf(fmaxf(s[0][r], s[1][r]), fmaxf(s[2][r], s[3][r]));
#pragma unroll
    for (int off = 1; off < 16; off <<= 1)
#pragma unroll
      for (int r = 0; r < 8; ++r)
        cm[r] = fmaxf(cm[r], __shfl_xor(cm[r], off, 32));

    float alpha[8];
#pragma unroll
    for (int r = 0; r < 8; ++r) {
      const float nm = fmaxf(mr[r], cm[r]);
      alpha[r] = __expf(mr[r] - nm);
      mr[r] = nm;
    }

    float rs[8];
#pragma unroll
    for (int r = 0; r < 8; ++r) rs[r] = 0.0f;
#pragma unroll
    for (int j = 0; j < 4; ++j)
#pragma unroll
      for (int r = 0; r < 8; ++r) {
        const float p = __expf(s[j][r] - mr[r]);
        rs[r] += p;
        const int rr = r + ((lane & 16) >> 1);
        Pl[rr * 72 + j * 16 + (lane & 15)] = (__bf16)p;  // C-layout -> LDS
      }
#pragma unroll
    for (int off = 1; off < 16; off <<= 1)
#pragma unroll
      for (int r = 0; r < 8; ++r) rs[r] += __shfl_xor(rs[r], off, 32);
#pragma unroll
    for (int r = 0; r < 8; ++r) lr[r] = lr[r] * alpha[r] + rs[r];
#pragma unroll
    for (int j = 0; j < 4; ++j)
#pragma unroll
      for (int r = 0; r < 8; ++r) o[j][r] *= alpha[r];

    __builtin_amdgcn_wave_barrier();   // order LDS P writes vs ds reads

    // ---- O += P @ V (contract over 64 s; V from transposed Vt) ----
#pragma unroll
    for (int kd = 0; kd < 2; ++kd) {
      const v16bf ap = load_frag(Pl, 72, 0, kd * 32);
#pragma unroll
      for (int j = 0; j < 4; ++j) {
        const v16bf bv = load_frag(Vp + s0, SKV, j * 16, kd * 32);
        o[j] = wmma_bf16(ap, bv, o[j]);
      }
    }
    __builtin_amdgcn_wave_barrier();
  }

  // ---- store unnormalized partials + (m, l) ----
  const size_t pbase = ((size_t)b * HEADS + h) * NSPLIT + split;
#pragma unroll
  for (int j = 0; j < 4; ++j)
#pragma unroll
    for (int r = 0; r < 8; ++r) {
      const int rr = row0 + r + ((lane & 16) >> 1);
      const int cc = j * 16 + (lane & 15);
      Opart[(pbase * NLAT + rr) * DHEAD + cc] = o[j][r];
    }
  if ((lane & 15) == 0) {
#pragma unroll
    for (int r = 0; r < 8; ++r) {
      const int rr = row0 + r + ((lane & 16) >> 1);
      mlbuf[pbase * NLAT * 2 + rr * 2 + 0] = mr[r];
      mlbuf[pbase * NLAT * 2 + rr * 2 + 1] = lr[r];
    }
  }
}

// ---------------------------------------------------------------------------
// Kernel 5: merge the NSPLIT flash partials -> Aout[b, n, h*64+d] bf16
// Grid (HEADS, BATCH), 256 threads: thread -> (row n, 16-wide d segment).
// ---------------------------------------------------------------------------
__global__ void __launch_bounds__(256)
attn_combine_kernel(const float* __restrict__ Opart,
                    const float* __restrict__ mlbuf,
                    __bf16* __restrict__ Aout) {
  const int h = blockIdx.x;
  const int b = blockIdx.y;
  const int t = threadIdx.x;
  const int n = t >> 2;              // 0..63
  const int dseg = (t & 3) * 16;
  const size_t base = ((size_t)b * HEADS + h) * NSPLIT;

  float mv[NSPLIT], lv[NSPLIT], M = -1e30f;
#pragma unroll
  for (int p = 0; p < NSPLIT; ++p) {
    mv[p] = mlbuf[(base + p) * NLAT * 2 + n * 2 + 0];
    lv[p] = mlbuf[(base + p) * NLAT * 2 + n * 2 + 1];
    M = fmaxf(M, mv[p]);
  }
  float L = 0.0f;
#pragma unroll
  for (int p = 0; p < NSPLIT; ++p) L += lv[p] * __expf(mv[p] - M);
  const float invL = 1.0f / L;

  float acc[16];
#pragma unroll
  for (int i = 0; i < 16; ++i) acc[i] = 0.0f;
#pragma unroll
  for (int p = 0; p < NSPLIT; ++p) {
    const float w = __expf(mv[p] - M);
    const float* op = Opart + ((base + p) * NLAT + n) * DHEAD + dseg;
#pragma unroll
    for (int i = 0; i < 16; ++i) acc[i] += w * op[i];
  }
#pragma unroll
  for (int i = 0; i < 16; ++i)
    Aout[((size_t)b * NLAT + n) * DIM + h * DHEAD + dseg + i] =
        (__bf16)(acc[i] * invL);
}

// ---------------------------------------------------------------------------
// Host-side launch
// ---------------------------------------------------------------------------
extern "C" void kernel_launch(void* const* d_in, const int* in_sizes, int n_in,
                              void* d_out, int out_size, void* d_ws,
                              size_t ws_size, hipStream_t stream) {
  (void)in_sizes; (void)n_in; (void)out_size; (void)ws_size;
  const float* x   = (const float*)d_in[0];
  const float* lat = (const float*)d_in[1];
  const float* g1  = (const float*)d_in[2];
  const float* b1  = (const float*)d_in[3];
  const float* g2  = (const float*)d_in[4];
  const float* b2  = (const float*)d_in[5];
  const float* Wq  = (const float*)d_in[6];
  const float* Wkv = (const float*)d_in[7];
  const float* Wo  = (const float*)d_in[8];
  float* out = (float*)d_out;

  char* ws = (char*)d_ws;
  auto take = [&](size_t bytes) -> char* {
    char* p = ws;
    ws += (bytes + 255) & ~(size_t)255;
    return p;
  };
  __bf16* kvin = (__bf16*)take((size_t)MKV * DIM * 2);          // ln(x)||ln(lat)
  __bf16* WqT  = (__bf16*)take((size_t)DIM * DIM * 2);
  __bf16* WkvT = (__bf16*)take((size_t)2 * DIM * DIM * 2);
  __bf16* WoT  = (__bf16*)take((size_t)DIM * DIM * 2);
  __bf16* Kbuf = (__bf16*)take((size_t)MKV * DIM * 2);          // [b,s,1024]
  __bf16* Vt   = (__bf16*)take((size_t)MKV * DIM * 2);          // [b,1024,s]
  __bf16* Qbuf = (__bf16*)take((size_t)BATCH * NLAT * DIM * 2); // [b,64,1024]
  __bf16* Aout = (__bf16*)take((size_t)BATCH * NLAT * DIM * 2); // attn out
  float*  Opart= (float*)take((size_t)BATCH * HEADS * NSPLIT * NLAT * DHEAD * 4);
  float*  mlb  = (float*)take((size_t)BATCH * HEADS * NSPLIT * NLAT * 2 * 4);

  // 1) LayerNorms -> bf16 concat buffer
  ln_to_bf16_kernel<<<BATCH * MSEQ, 256, 0, stream>>>(x, g1, b1, kvin, MSEQ, 0);
  ln_to_bf16_kernel<<<BATCH * NLAT, 256, 0, stream>>>(lat, g2, b2, kvin, NLAT,
                                                      MSEQ);
  // 2) Weight transposes -> bf16
  transpose_to_bf16_kernel<<<dim3(DIM / 32, DIM / 32), dim3(32, 8), 0, stream>>>(
      Wq, WqT, DIM, DIM);
  transpose_to_bf16_kernel<<<dim3(2 * DIM / 32, DIM / 32), dim3(32, 8), 0,
                             stream>>>(Wkv, WkvT, DIM, 2 * DIM);
  transpose_to_bf16_kernel<<<dim3(DIM / 32, DIM / 32), dim3(32, 8), 0, stream>>>(
      Wo, WoT, DIM, DIM);

  // 3) KV projection (dominant GEMM): [33280,1024] x [1024,2048]
  gemm_wmma_kernel<2><<<dim3(MKV / 128, (2 * DIM) / 128), 256, 0, stream>>>(
      kvin, WkvT, DIM, DIM, MKV, (size_t)0, Kbuf, Vt, 0);

  // 4) Q projection: ln rows live inside kvin at offset MSEQ per batch
  gemm_wmma_kernel<0><<<dim3((BATCH * NLAT) / 128, DIM / 128), 256, 0, stream>>>(
      kvin + (size_t)MSEQ * DIM, WqT, DIM, DIM, NLAT, (size_t)SKV * DIM, Qbuf,
      nullptr, DIM);

  // 5) Flash attention partials (s split 5 ways) + combine
  attn_wmma_kernel<<<dim3(HEADS, BATCH, NSPLIT), 128, 0, stream>>>(
      Qbuf, Kbuf, Vt, Opart, mlb);
  attn_combine_kernel<<<dim3(HEADS, BATCH), 256, 0, stream>>>(Opart, mlb, Aout);

  // 6) Output projection -> f32 d_out
  gemm_wmma_kernel<1><<<dim3((BATCH * NLAT) / 128, DIM / 128), 256, 0, stream>>>(
      Aout, WoT, DIM, DIM, BATCH * NLAT, (size_t)0, out, nullptr, DIM);
}
